// esem_41248865910988
// MI455X (gfx1250) — compile-verified, matrix operand order checked by reading, not run on hardware
//
#include <hip/hip_runtime.h>

typedef unsigned short u16;
typedef __attribute__((ext_vector_type(16))) __bf16 v16bf;
typedef __attribute__((ext_vector_type(8)))  float  v8f;
typedef __attribute__((ext_vector_type(8)))  unsigned short us8;

__device__ __forceinline__ u16 f2bf_rne(float f) {
  unsigned int u = __float_as_uint(f);
  u += 0x7FFFu + ((u >> 16) & 1u);
  return (u16)(u >> 16);
}
__device__ __forceinline__ float bf2f(u16 h) {
  return __uint_as_float((unsigned int)h << 16);
}

union BfFrag { v16bf v; us8 h[2]; };

__device__ __forceinline__ v8f wmma_bf16(const BfFrag& a, const BfFrag& b, v8f c) {
  return __builtin_amdgcn_wmma_f32_16x16x32_bf16(false, a.v, false, b.v,
                                                 (short)0, c, false, false);
}

// ---------------------------------------------------------------------------
// WMMA GEMM, bf16 operands, fp32 accumulate. One wave computes a 32x64 tile
// (2 row-tiles x 4 col-tiles => 8 wmmas per 32-K step, fed by 12 b128 loads).
// Double-buffered fragments (even/odd banks) so loads overlap the wmmas.
// Requires M%32==0, N%64==0, K%32==0 (true for every GEMM in this net).
//   A: bf16 row-major (M,K); lane l (half=l>>4, r=l&15) supplies row r of its
//      row-tile, element e -> K = (e&7)+8*half+16*(e>=8): two b128 runs.
//   P: weights pre-packed in fragment order: two contiguous b128 per tile.
//   Epilogue: (acc+bias)*scale+shift, optional ReLU; writes fp32 and/or bf16.
// ---------------------------------------------------------------------------
__global__ __launch_bounds__(32)
void wmma_gemm_kernel(const u16* __restrict__ A, const u16* __restrict__ P,
                      const float* __restrict__ bias,
                      const float* __restrict__ scale, const float* __restrict__ shift,
                      float* __restrict__ outF, u16* __restrict__ outB,
                      int M, int N, int K, int relu) {
  const int lane = threadIdx.x & 31;
  const int half = lane >> 4;
  const int r    = lane & 15;
  const int m0   = blockIdx.x * 32;
  const int n0   = blockIdx.y * 64;
  const int Kt   = K >> 5;

  const us8* Ap0 = (const us8*)(A + (size_t)(m0 + r) * K + 8 * half);
  const us8* Ap1 = (const us8*)(A + (size_t)(m0 + 16 + r) * K + 8 * half);
  const size_t bstride = (size_t)Kt * 512;       // elements per packed n-tile
  const us8* Bp0 = (const us8*)(P + (size_t)(blockIdx.y * 4 + 0) * bstride + lane * 16);
  const us8* Bp1 = (const us8*)(P + (size_t)(blockIdx.y * 4 + 1) * bstride + lane * 16);
  const us8* Bp2 = (const us8*)(P + (size_t)(blockIdx.y * 4 + 2) * bstride + lane * 16);
  const us8* Bp3 = (const us8*)(P + (size_t)(blockIdx.y * 4 + 3) * bstride + lane * 16);

  BfFrag aE[2], bE[4], aO[2], bO[4];
  v8f acc[2][4];
#pragma unroll
  for (int rt = 0; rt < 2; ++rt)
#pragma unroll
    for (int ct = 0; ct < 4; ++ct)
      acc[rt][ct] = v8f{0.f, 0.f, 0.f, 0.f, 0.f, 0.f, 0.f, 0.f};

  auto loadBank = [&](BfFrag (&aa)[2], BfFrag (&bb)[4]) {
    aa[0].h[0] = Ap0[0]; aa[0].h[1] = Ap0[2];
    aa[1].h[0] = Ap1[0]; aa[1].h[1] = Ap1[2];
    bb[0].h[0] = Bp0[0]; bb[0].h[1] = Bp0[1];
    bb[1].h[0] = Bp1[0]; bb[1].h[1] = Bp1[1];
    bb[2].h[0] = Bp2[0]; bb[2].h[1] = Bp2[1];
    bb[3].h[0] = Bp3[0]; bb[3].h[1] = Bp3[1];
    Ap0 += 4; Ap1 += 4;                // +32 K elements
    Bp0 += 64; Bp1 += 64; Bp2 += 64; Bp3 += 64;   // next packed k-tile
  };
  auto wmmaBank = [&](BfFrag (&aa)[2], BfFrag (&bb)[4]) {
#pragma unroll
    for (int rt = 0; rt < 2; ++rt)
#pragma unroll
      for (int ct = 0; ct < 4; ++ct)
        acc[rt][ct] = wmma_bf16(aa[rt], bb[ct], acc[rt][ct]);
  };

  loadBank(aE, bE);
  int kt = 0;
  for (; kt + 1 < Kt; kt += 2) {
    loadBank(aO, bO);
    __builtin_prefetch((const void*)Ap0, 0, 3);   // speculative stream prefetch
    __builtin_prefetch((const void*)Ap1, 0, 3);
    wmmaBank(aE, bE);
    if (kt + 2 < Kt) loadBank(aE, bE);
    wmmaBank(aO, bO);
  }
  if (kt < Kt) wmmaBank(aE, bE);       // odd-Kt tail (last loaded bank is E)

  // ---- epilogue ----
#pragma unroll
  for (int ct = 0; ct < 4; ++ct) {
    const int col = n0 + ct * 16 + r;
    const float bb = bias[col];
    const float sc = scale ? scale[col] : 1.f;
    const float sh = scale ? shift[col] : 0.f;
#pragma unroll
    for (int rt = 0; rt < 2; ++rt) {
#pragma unroll
      for (int e = 0; e < 8; ++e) {
        int mo = m0 + rt * 16 + e + 8 * half;
        float v = (acc[rt][ct][e] + bb) * sc + sh;
        if (relu) v = fmaxf(v, 0.f);
        size_t oi = (size_t)mo * N + col;
        if (outF) outF[oi] = v;
        if (outB) outB[oi] = f2bf_rne(v);
      }
    }
  }
}

// ---------------------------------------------------------------------------
// Weight pack: W (K,N) fp32 row-major -> bf16 fragment order.
// packed[((nt*Kt+kt)*32 + lane)*16 + e] = bf16(W[k*N+n]),
//   k = kt*32 + (e&7) + 8*(lane>>4) + 16*(e>=8), n = nt*16 + (lane&15)
// ---------------------------------------------------------------------------
__global__ void pack_w_kernel(const float* __restrict__ W, u16* __restrict__ P,
                              int K, int N) {
  long long idx = (long long)blockIdx.x * blockDim.x + threadIdx.x;
  if (idx >= (long long)K * N) return;
  int e    = (int)(idx & 15);
  int lane = (int)((idx >> 4) & 31);
  int t    = (int)(idx >> 9);          // nt*Kt + kt
  int Kt = K >> 5;
  int kt = t % Kt, ntile = t / Kt;
  int k = kt * 32 + (e & 7) + 8 * (lane >> 4) + ((e >= 8) ? 16 : 0);
  int n = ntile * 16 + (lane & 15);
  P[idx] = f2bf_rne(W[(size_t)k * N + n]);
}

// ---------------------------------------------------------------------------
// Direct 3x3 stride-2 VALID conv, fused bias + BN + ReLU, NCHW, fp32 math.
// mode 0: (N,Cin,H,W) contiguous; mode 1: n -> x[n>>4, chOff+(n&15)] (Cin=1);
// mode 2: ci -> x[n, 16+ci]. Output fp32 (outF) or bf16 (outB).
// ---------------------------------------------------------------------------
__global__ void conv3x3s2_kernel(const float* __restrict__ in,
                                 const float* __restrict__ W,
                                 const float* __restrict__ bias,
                                 const float* __restrict__ scale,
                                 const float* __restrict__ shift,
                                 float* __restrict__ outF, u16* __restrict__ outB,
                                 int N, int Cin, int Hin, int Hout,
                                 int mode, int chOff) {
  long long idx = (long long)blockIdx.x * blockDim.x + threadIdx.x;
  long long total = (long long)N * 32 * Hout * Hout;
  if (idx >= total) return;
  int wo = (int)(idx % Hout);
  int ho = (int)((idx / Hout) % Hout);
  int co = (int)((idx / ((long long)Hout * Hout)) % 32);
  int n  = (int)(idx / ((long long)Hout * Hout * 32));
  const size_t HW = (size_t)Hin * Hin;
  float acc = 0.f;
  for (int ci = 0; ci < Cin; ++ci) {
    size_t base;
    if (mode == 1)      base = ((size_t)(n >> 4) * 32 + (n & 15) + chOff) * HW;
    else if (mode == 2) base = ((size_t)n * 32 + 16 + ci) * HW;
    else                base = ((size_t)n * Cin + ci) * HW;
    const float* ip = in + base + (size_t)(2 * ho) * Hin + 2 * wo;
    const float* wp = W + (size_t)(co * Cin + ci) * 9;
#pragma unroll
    for (int kh = 0; kh < 3; ++kh)
#pragma unroll
      for (int kw = 0; kw < 3; ++kw)
        acc = __builtin_fmaf(ip[(size_t)kh * Hin + kw], wp[kh * 3 + kw], acc);
  }
  acc = fmaxf((acc + bias[co]) * scale[co] + shift[co], 0.f);
  if (outB) outB[idx] = f2bf_rne(acc);
  else      outF[idx] = acc;
}

// ---------------------------------------------------------------------------
// Index tables (REG_TRI 52x3, ANS_REG 26x2), generated on-device.
// ---------------------------------------------------------------------------
__global__ void build_tables_kernel(int* __restrict__ regtri, int* __restrict__ ansreg) {
  if (threadIdx.x != 0 || blockIdx.x != 0) return;
  int c = 0;
  for (int i = 0; i < 8; ++i)
    for (int j = 0; j < i; ++j)
      for (int k = 0; k < j; ++k) {
        int a = 7 - i, b = 7 - j, d = 7 - k;
        bool sp = (a == 0 && b == 1 && d == 2) || (a == 3 && b == 4 && d == 5) ||
                  (a == 0 && b == 3 && d == 6) || (a == 1 && b == 4 && d == 7);
        if (!sp) { regtri[c * 3] = a; regtri[c * 3 + 1] = b; regtri[c * 3 + 2] = d; ++c; }
      }
  c = 0;
  for (int i = 0; i < 8; ++i)
    for (int j = 0; j < i; ++j) {
      int a = 7 - i, b = 7 - j;
      if (!((a == 2 && b == 5) || (a == 6 && b == 7))) {
        ansreg[c * 2] = a; ansreg[c * 2 + 1] = b; ++c;
      }
    }
}

// ---------------------------------------------------------------------------
// Gather / concat kernels -> bf16 GEMM inputs. PE (512,256): row b*16+p.
// ---------------------------------------------------------------------------
__global__ void build_fusin_kernel(const float* __restrict__ EL, const float* __restrict__ ES,
                                   u16* __restrict__ Z) {
  int idx = blockIdx.x * blockDim.x + threadIdx.x;
  if (idx >= 512 * 512) return;
  int row = idx >> 9, c = idx & 511;
  float v = (c < 256) ? EL[row * 256 + c] : ES[row * 256 + (c - 256)];
  Z[idx] = f2bf_rne(v);
}

__global__ void build_z1_kernel(const float* __restrict__ PE, const float* __restrict__ PE8,
                                const int* __restrict__ regtri, u16* __restrict__ Z) {
  long long idx = (long long)blockIdx.x * blockDim.x + threadIdx.x;
  if (idx >= 1728LL * 1024) return;
  int c = (int)(idx & 1023); int row = (int)(idx >> 10);
  int b = row / 54, n = row % 54;
  float v;
  if (c >= 768)     v = PE8[b * 256 + (c - 768)];
  else if (n >= 52) v = 0.f;
  else {
    int p = regtri[n * 3 + (c >> 8)];
    v = PE[(b * 16 + p) * 256 + (c & 255)];
  }
  Z[idx] = f2bf_rne(v);
}

__global__ void build_z2_kernel(const float* __restrict__ PE, const float* __restrict__ PE8,
                                u16* __restrict__ Z) {
  const int tri[4][3] = {{0, 3, 6}, {1, 4, 7}, {0, 1, 2}, {3, 4, 5}};
  long long idx = (long long)blockIdx.x * blockDim.x + threadIdx.x;
  if (idx >= 128LL * 1024) return;
  int c = (int)(idx & 1023); int row = (int)(idx >> 10);
  int b = row >> 2, t = row & 3;
  float v = (c >= 768) ? PE8[b * 256 + (c - 768)]
                       : PE[(b * 16 + tri[t][c >> 8]) * 256 + (c & 255)];
  Z[idx] = f2bf_rne(v);
}

__global__ void build_z3_kernel(const float* __restrict__ PE, const float* __restrict__ PE8,
                                const int* __restrict__ ansreg, u16* __restrict__ Z) {
  long long idx = (long long)blockIdx.x * blockDim.x + threadIdx.x;
  if (idx >= 6656LL * 1024) return;
  int c = (int)(idx & 1023); int row = (int)(idx >> 10);
  int n = row % 26; int ba = row / 26; int a = ba & 7; int b = ba >> 3;
  float v;
  if (c < 512) {
    int p = ansreg[n * 2 + (c >> 8)];
    v = PE[(b * 16 + p) * 256 + (c & 255)];
  } else if (c < 768) v = PE[(b * 16 + 8 + a) * 256 + (c - 512)];
  else                v = PE8[b * 256 + (c - 768)];
  Z[idx] = f2bf_rne(v);
}

__global__ void build_z45_kernel(const float* __restrict__ PE, const float* __restrict__ PE8,
                                 u16* __restrict__ Z) {
  long long idx = (long long)blockIdx.x * blockDim.x + threadIdx.x;
  if (idx >= 512LL * 1024) return;
  int c = (int)(idx & 1023); int row = (int)(idx >> 10);
  int s = row >> 8; int ba = row & 255; int b = ba >> 3; int a = ba & 7;
  int p0 = s ? 6 : 2, p1 = s ? 7 : 5;
  float v;
  if (c < 256)      v = PE[(b * 16 + p0) * 256 + c];
  else if (c < 512) v = PE[(b * 16 + p1) * 256 + (c - 256)];
  else if (c < 768) v = PE[(b * 16 + 8 + a) * 256 + (c - 512)];
  else              v = PE8[b * 256 + (c - 768)];
  Z[idx] = f2bf_rne(v);
}

// ---------------------------------------------------------------------------
// Reductions / tail
// ---------------------------------------------------------------------------
__global__ void reduce_ctx_kernel(const float* __restrict__ Z1o, const float* __restrict__ Z2o,
                                  float* __restrict__ CTXG) {
  int idx = blockIdx.x * blockDim.x + threadIdx.x;
  if (idx >= 32 * 512) return;
  int b = idx >> 9, c = idx & 511;
  float s = 0.f;
  for (int n = 0; n < 54; ++n) s += Z1o[(size_t)(b * 54 + n) * 512 + c];
  for (int t = 0; t < 4;  ++t) s += Z2o[(size_t)(b * 4 + t) * 512 + c];
  CTXG[idx] = s;
}

__global__ void gout_kernel(const float* __restrict__ CTXG, const float* __restrict__ Z3o,
                            const float* __restrict__ Z45o, u16* __restrict__ GOUTb) {
  int idx = blockIdx.x * blockDim.x + threadIdx.x;
  if (idx >= 256 * 512) return;
  int row = idx >> 9, c = idx & 511; int b = row >> 3;
  float s = CTXG[b * 512 + c];
  for (int n = 0; n < 26; ++n) s += Z3o[(size_t)(row * 26 + n) * 512 + c];
  s += Z45o[(size_t)row * 512 + c];
  s += Z45o[(size_t)(256 + row) * 512 + c];
  GOUTb[idx] = f2bf_rne(s);
}

// Final f-layer: (256,256) @ (256,1) + b  — plain dot, WMMA is wasted on N=1.
__global__ void fdot_kernel(const u16* __restrict__ F2b, const float* __restrict__ W,
                            const float* __restrict__ b, float* __restrict__ out) {
  int row = blockIdx.x * blockDim.x + threadIdx.x;
  if (row >= 256) return;
  float s = b[0];
  for (int k = 0; k < 256; ++k)
    s = __builtin_fmaf(bf2f(F2b[row * 256 + k]), W[k], s);
  out[row] = s;
}

__global__ void logsoftmax_kernel(const float* __restrict__ f, float* __restrict__ out) {
  int b = threadIdx.x;
  if (blockIdx.x != 0 || b >= 32) return;
  float v[8]; float m = -1e30f;
  for (int a = 0; a < 8; ++a) { v[a] = f[b * 8 + a]; m = fmaxf(m, v[a]); }
  float s = 0.f;
  for (int a = 0; a < 8; ++a) s += expf(v[a] - m);
  float ls = logf(s);
  for (int a = 0; a < 8; ++a) out[b * 8 + a] = v[a] - m - ls;
}

// ---------------------------------------------------------------------------
// Host-side orchestration
// ---------------------------------------------------------------------------
static inline unsigned blocks1(long long total, int bs) {
  return (unsigned)((total + bs - 1) / bs);
}

static void gemm(const u16* A, const u16* P, const float* bias, const float* scale,
                 const float* shift, float* outF, u16* outB,
                 int M, int N, int K, int relu, hipStream_t s) {
  dim3 grid((unsigned)(M / 32), (unsigned)(N / 64));
  wmma_gemm_kernel<<<grid, 32, 0, s>>>(A, P, bias, scale, shift, outF, outB, M, N, K, relu);
}

static void conv(const float* in, const float* W, const float* b, const float* sc,
                 const float* sh, float* outF, u16* outB, int N, int Cin, int Hin,
                 int Hout, int mode, int chOff, hipStream_t s) {
  long long total = (long long)N * 32 * Hout * Hout;
  conv3x3s2_kernel<<<blocks1(total, 256), 256, 0, s>>>(in, W, b, sc, sh, outF, outB,
                                                       N, Cin, Hin, Hout, mode, chOff);
}

extern "C" void kernel_launch(void* const* d_in, const int* in_sizes, int n_in,
                              void* d_out, int out_size, void* d_ws, size_t ws_size,
                              hipStream_t stream) {
  (void)in_sizes; (void)n_in; (void)out_size; (void)ws_size;
  auto in = [&](int i) { return (const float*)d_in[i]; };

  // Input indices: recursive insertion-order flatten of setup_inputs().
  // 0:x  1..16:cnn_shape  17..32:cnn_line  33..48:cnn_global (W0,b0..W3,b3,
  // bn scale/shift x4)  49..64:pre  65..80:g  81..96:g2  97..106:f
  const float* x = in(0);

  // ---- workspace (byte bump allocator, 16B aligned) ----
  char* base = (char*)d_ws;
  size_t o = 0;
  auto allocB = [&](size_t bytes) { void* p = base + o; o = (o + bytes + 15) & ~15ull; return p; };
  auto allocF = [&](size_t n) { return (float*)allocB(n * 4); };
  auto allocH = [&](size_t n) { return (u16*)allocB(n * 2); };

  int* regtri = (int*)allocB(208 * 4);
  int* ansreg = regtri + 156;
  float* bufA = allocF(102252544ull);  // 512*32*79*79
  float* bufB = allocF(24920064ull);   // 512*32*39*39
  float* bufC = allocF(5914624ull);    // 512*32*19*19
  u16* SFb    = allocH(512ull * 2592);
  u16* LFb    = allocH(512ull * 2592);
  u16* GEb    = allocH(32ull * 2592);
  float* PE8  = allocF(32ull * 256);
  float* ES   = allocF(512ull * 256);
  float* EL   = allocF(512ull * 256);
  u16* FUSINb = allocH(512ull * 512);
  float* PE   = allocF(512ull * 256);
  // packed weights (bf16, fragment order)
  u16* Pfc_shape = allocH(2592ull * 256);
  u16* Pfc_line  = allocH(2592ull * 256);
  u16* Pfc2      = allocH(2592ull * 256);
  u16* Pfusion   = allocH(512ull * 256);
  const int gdims[5] = {1024, 1024, 768, 512, 512};
  u16* Pg[4];  for (int l = 0; l < 4; ++l) Pg[l]  = allocH((size_t)gdims[l] * gdims[l + 1]);
  u16* Pg2[4]; for (int l = 0; l < 4; ++l) Pg2[l] = allocH((size_t)gdims[l] * gdims[l + 1]);
  u16* Pf0 = allocH(512ull * 256);
  u16* Pf1 = allocH(256ull * 256);

  // Z-stage buffers alias bufA (conv stage finished before they are used)
  char* zb = (char*)bufA; size_t zo = 0;
  auto zallocB = [&](size_t bytes) { void* p = zb + zo; zo = (zo + bytes + 15) & ~15ull; return p; };
  auto zF = [&](size_t n) { return (float*)zallocB(n * 4); };
  auto zH = [&](size_t n) { return (u16*)zallocB(n * 2); };
  u16* Z3INb = zH(6656ull * 1024); u16* Z3Ab = zH(6656ull * 1024); u16* Z3Bb = zH(6656ull * 1024);
  float* Z3o = zF(6656ull * 512);
  u16* Z1INb = zH(1728ull * 1024); u16* Z1Ab = zH(1728ull * 1024); u16* Z1Bb = zH(1728ull * 1024);
  float* Z1o = zF(1728ull * 512);
  u16* Z2INb = zH(128ull * 1024);  u16* Z2Ab = zH(128ull * 1024);  u16* Z2Bb = zH(128ull * 1024);
  float* Z2o = zF(128ull * 512);
  u16* Z45INb = zH(512ull * 1024); u16* Z45Ab = zH(512ull * 1024); u16* Z45Bb = zH(512ull * 1024);
  float* Z45o = zF(512ull * 512);
  float* CTXG = zF(32ull * 512);
  u16* GOUTb  = zH(256ull * 512);
  u16* F1b    = zH(256ull * 256);
  u16* F2b    = zH(256ull * 256);
  float* FOUT = zF(256);

  build_tables_kernel<<<1, 32, 0, stream>>>(regtri, ansreg);

  // ---- pack all GEMM weights once ----
  auto pack = [&](const float* W, u16* P, int K, int N) {
    pack_w_kernel<<<blocks1((long long)K * N, 256), 256, 0, stream>>>(W, P, K, N);
  };
  pack(in(49), Pfc_shape, 2592, 256);
  pack(in(53), Pfc_line,  2592, 256);
  pack(in(57), Pfc2,      2592, 256);
  pack(in(61), Pfusion,   512, 256);
  for (int l = 0; l < 4; ++l) pack(in(65 + 2 * l), Pg[l],  gdims[l], gdims[l + 1]);
  for (int l = 0; l < 4; ++l) pack(in(81 + 2 * l), Pg2[l], gdims[l], gdims[l + 1]);
  pack(in(97), Pf0, 512, 256);
  pack(in(99), Pf1, 256, 256);

  // ---- CNN chains (base cb: W=cb+2l, b=cb+2l+1, scale=cb+8+2l, shift=cb+9+2l) ----
  auto cnn4 = [&](int cb, int N, int Cin0, int mode, int chOff, u16* resultB) {
    conv(x,    in(cb + 0), in(cb + 1), in(cb + 8),  in(cb + 9),  bufA, nullptr, N, Cin0, 160, 79, mode, chOff, stream);
    conv(bufA, in(cb + 2), in(cb + 3), in(cb + 10), in(cb + 11), bufB, nullptr, N, 32, 79, 39, 0, 0, stream);
    conv(bufB, in(cb + 4), in(cb + 5), in(cb + 12), in(cb + 13), bufC, nullptr, N, 32, 39, 19, 0, 0, stream);
    conv(bufC, in(cb + 6), in(cb + 7), in(cb + 14), in(cb + 15), nullptr, resultB, N, 32, 19, 9, 0, 0, stream);
  };
  cnn4(1,  512, 1,  1, 0,  SFb);   // cnn_shape on x[:, :16]
  cnn4(17, 512, 1,  1, 16, LFb);   // cnn_line  on x[:, 16:]
  cnn4(33, 32,  16, 2, 0,  GEb);   // cnn_global on x[:, 16:]

  // ---- heads ----
  gemm(GEb, Pfc2,      in(58), in(59), in(60), PE8, nullptr, 32, 256, 2592, 1, stream);
  gemm(SFb, Pfc_shape, in(50), in(51), in(52), ES,  nullptr, 512, 256, 2592, 0, stream);
  gemm(LFb, Pfc_line,  in(54), in(55), in(56), EL,  nullptr, 512, 256, 2592, 0, stream);
  build_fusin_kernel<<<blocks1(512LL * 512, 256), 256, 0, stream>>>(EL, ES, FUSINb);
  gemm(FUSINb, Pfusion, in(62), in(63), in(64), PE, nullptr, 512, 256, 512, 1, stream);

  // ---- g-MLP chains: 1024 -> 1024 -> 768 -> 512 -> 512, bn+relu each ----
  auto run_g = [&](const u16* Zin, u16* bA, u16* bB, float* outF, int rows,
                   u16* const* Pw, int gb) {
    const u16* cur = Zin;
    u16* outs[3] = {bA, bB, bA};
    for (int l = 0; l < 4; ++l) {
      bool last = (l == 3);
      gemm(cur, Pw[l], in(gb + 2 * l + 1), in(gb + 8 + 2 * l), in(gb + 9 + 2 * l),
           last ? outF : nullptr, last ? nullptr : outs[l],
           rows, gdims[l + 1], gdims[l], 1, stream);
      if (!last) cur = outs[l];
    }
  };

  build_z1_kernel<<<blocks1(1728LL * 1024, 256), 256, 0, stream>>>(PE, PE8, regtri, Z1INb);
  run_g(Z1INb, Z1Ab, Z1Bb, Z1o, 1728, Pg2, 81);
  build_z2_kernel<<<blocks1(128LL * 1024, 256), 256, 0, stream>>>(PE, PE8, Z2INb);
  run_g(Z2INb, Z2Ab, Z2Bb, Z2o, 128, Pg, 65);
  reduce_ctx_kernel<<<blocks1(32LL * 512, 256), 256, 0, stream>>>(Z1o, Z2o, CTXG);

  build_z3_kernel<<<blocks1(6656LL * 1024, 256), 256, 0, stream>>>(PE, PE8, ansreg, Z3INb);
  run_g(Z3INb, Z3Ab, Z3Bb, Z3o, 6656, Pg2, 81);
  build_z45_kernel<<<blocks1(512LL * 1024, 256), 256, 0, stream>>>(PE, PE8, Z45INb);
  run_g(Z45INb, Z45Ab, Z45Bb, Z45o, 512, Pg, 65);

  gout_kernel<<<blocks1(256LL * 512, 256), 256, 0, stream>>>(CTXG, Z3o, Z45o, GOUTb);

  // ---- f-MLP: 512 -> 256 (bn,relu) -> 256 (bn,relu) -> 1 ----
  gemm(GOUTb, Pf0, in(98),  in(103), in(104), nullptr, F1b, 256, 256, 512, 1, stream);
  gemm(F1b,   Pf1, in(100), in(105), in(106), nullptr, F2b, 256, 256, 256, 1, stream);
  fdot_kernel<<<1, 256, 0, stream>>>(F2b, in(101), in(102), FOUT);

  logsoftmax_kernel<<<1, 32, 0, stream>>>(FOUT, (float*)d_out);
}
